// TJDLayer_80814104641708
// MI455X (gfx1250) — compile-verified, matrix operand order checked by reading, not run on hardware
//
#include <hip/hip_runtime.h>
#include <math.h>

#define B_ 128
#define T_ 8
#define D_ 2048
#define R_ 4
#define V_ 512
#define N_ (V_ * R_ * R_)   // 8192; col n = r1*(V_*R_) + v*R_ + r2
#define BN_ 64              // N-tile per block (4 WMMA sub-tiles per wave)

typedef _Float16 v16h __attribute__((ext_vector_type(16)));
typedef _Float16 v8h  __attribute__((ext_vector_type(8)));
typedef float    v8f  __attribute__((ext_vector_type(8)));

// ---------------- Kernel 1: fold the T-mean (linear, commutes with matmul) ----
__global__ void mean_embs_kernel(const float* __restrict__ x,
                                 float* __restrict__ mean32,
                                 _Float16* __restrict__ mean16) {
  int idx = blockIdx.x * blockDim.x + threadIdx.x;
  if (idx >= B_ * D_) return;
  int b = idx / D_, d = idx % D_;
  const float* p = x + (size_t)b * T_ * D_ + d;
  float s = 0.f;
#pragma unroll
  for (int t = 0; t < T_; ++t) s += p[(size_t)t * D_];
  s *= (1.0f / T_);
  mean32[idx] = s;
  mean16[idx] = (_Float16)s;
}

// ---------------- Kernel 2: alpha/beta (N=4 — plain dot products) ------------
__global__ void alphabeta_kernel(const float* __restrict__ mean32,
                                 const float* __restrict__ wa,
                                 const float* __restrict__ wb,
                                 float* __restrict__ alpha,
                                 float* __restrict__ beta) {
  __shared__ float red[256][8];
  const int b = blockIdx.x, t = threadIdx.x;
  float acc[8] = {0.f, 0.f, 0.f, 0.f, 0.f, 0.f, 0.f, 0.f};
  for (int d = t; d < D_; d += 256) {
    float m = mean32[b * D_ + d];
#pragma unroll
    for (int r = 0; r < R_; ++r) {
      acc[r]     += m * wa[d * R_ + r];
      acc[4 + r] += m * wb[d * R_ + r];
    }
  }
#pragma unroll
  for (int i = 0; i < 8; ++i) red[t][i] = acc[i];
  __syncthreads();
  for (int s = 128; s > 0; s >>= 1) {
    if (t < s) {
#pragma unroll
      for (int i = 0; i < 8; ++i) red[t][i] += red[t + s][i];
    }
    __syncthreads();
  }
  if (t < R_) {
    alpha[b * R_ + t] = fabsf(red[0][t]);
    beta[b * R_ + t]  = fabsf(red[0][4 + t]);
  }
}

// ---------------- Kernel 3: core = |mean_embs @ w_vocab| via WMMA f16 --------
// Block = 256 threads = 8 waves covering full M=128; block owns a 64-wide
// N strip (4 sub-tiles of 16 per wave -> 4 WMMAs re-use one A fragment).
// B chunk [32k x 64n] staged in LDS as [n][k] f16: each thread owns one n
// column and 8 consecutive k (8 coalesced b32 loads -> one ds_store_b128).
__global__ void core_gemm_kernel(const _Float16* __restrict__ A,   // [128,2048] f16
                                 const float* __restrict__ Wv,     // [2048,8192] f32
                                 float* __restrict__ core) {       // [128,8192] f32 (abs)
  __shared__ _Float16 ldsB[BN_][32];   // [n][k]; row = 64B, 16B-aligned v8h slices
  const int n0   = blockIdx.x * BN_;
  const int tid  = threadIdx.x;
  const int wave = tid >> 5;
  const int lane = tid & 31;
  const int m0   = wave * 16;
  const int grp  = lane >> 4;          // half-wave (0/1)
  const int lr   = lane & 15;
  const int m    = m0 + lr;            // A row for this lane

  // staging coordinates: one n column, 8 consecutive k per thread
  const int snn  = tid & 63;           // 0..63
  const int skk0 = (tid >> 6) * 8;     // {0,8,16,24}

  v8f c0 = {}, c1 = {}, c2 = {}, c3 = {};

  for (int k0 = 0; k0 < D_; k0 += 32) {
    __syncthreads();   // LDS reads of previous chunk done
    {
      const float* wp = Wv + (size_t)(k0 + skk0) * N_ + n0 + snn;
      v8h h;
#pragma unroll
      for (int i = 0; i < 8; ++i) h[i] = (_Float16)wp[(size_t)i * N_];
      *(v8h*)&ldsB[snn][skk0] = h;
      if (k0 + 32 < D_)  // prefetch next chunk (speculative, RT)
        __builtin_prefetch(wp + (size_t)32 * N_, 0, 0);
    }
    __syncthreads();

    // A fragment: ISA 16-bit A 16x32 layout.
    // lane<16:  h0..7 -> K=k0..+7,    h8..15 -> K=k0+16..+23
    // lane>=16: h0..7 -> K=k0+8..+15, h8..15 -> K=k0+24..+31
    const _Float16* arow = A + (size_t)m * D_ + k0 + grp * 8;
    v8h alo = *(const v8h*)(arow);
    v8h ahi = *(const v8h*)(arow + 16);
    v16h a;
#pragma unroll
    for (int i = 0; i < 8; ++i) { a[i] = alo[i]; a[8 + i] = ahi[i]; }

    // 4 B fragments (N = s*16 + lane%16) from LDS; A re-used across all 4
#pragma unroll
    for (int s = 0; s < 4; ++s) {
      const _Float16* brow = &ldsB[s * 16 + lr][grp * 8];
      v8h blo = *(const v8h*)(brow);
      v8h bhi = *(const v8h*)(brow + 16);
      v16h bb;
#pragma unroll
      for (int i = 0; i < 8; ++i) { bb[i] = blo[i]; bb[8 + i] = bhi[i]; }
      v8f* cp = (s == 0) ? &c0 : (s == 1) ? &c1 : (s == 2) ? &c2 : &c3;
      *cp = __builtin_amdgcn_wmma_f32_16x16x32_f16(false, a, false, bb,
                                                   (short)0, *cp, false, false);
    }
  }

  // C layout: VGPR j -> M = m0 + j + 8*(lane>=16), N = n0 + s*16 + lane%16
  const int mbase = m0 + (grp ? 8 : 0);
#pragma unroll
  for (int s = 0; s < 4; ++s) {
    const v8f c = (s == 0) ? c0 : (s == 1) ? c1 : (s == 2) ? c2 : c3;
    const int n = n0 + s * 16 + lr;
#pragma unroll
    for (int j = 0; j < 8; ++j)
      core[(size_t)(mbase + j) * N_ + n] = fabsf(c[j]);
  }
}

// ---------------- Kernel 4: per-batch TT epilogue ----------------------------
__global__ void loss_kernel(const float* __restrict__ core,
                            const float* __restrict__ alpha,
                            const float* __restrict__ beta,
                            const int* __restrict__ labels,
                            float* __restrict__ loss_b) {
  __shared__ float red[256][16];
  const int b = blockIdx.x, t = threadIdx.x;
  const float* cb = core + (size_t)b * N_;

  // marg[r1][r2] = sum_v core[b, r1, v, r2]
  float acc[16];
#pragma unroll
  for (int i = 0; i < 16; ++i) acc[i] = 0.f;
  for (int v = t; v < V_; v += 256) {
#pragma unroll
    for (int i = 0; i < R_; ++i)
#pragma unroll
      for (int j = 0; j < R_; ++j)
        acc[i * R_ + j] += cb[i * (V_ * R_) + v * R_ + j];
  }
#pragma unroll
  for (int i = 0; i < 16; ++i) red[t][i] = acc[i];
  __syncthreads();
  for (int s = 128; s > 0; s >>= 1) {
    if (t < s) {
#pragma unroll
      for (int i = 0; i < 16; ++i) red[t][i] += red[t + s][i];
    }
    __syncthreads();
  }

  if (t == 0) {
    float marg[R_][R_];
#pragma unroll
    for (int i = 0; i < R_; ++i)
#pragma unroll
      for (int j = 0; j < R_; ++j) marg[i][j] = red[0][i * R_ + j];

    // chain of selected 4x4 cores:  res = M_0 @ M_1 @ ... @ M_{T-1}
    float res[R_][R_];
    {
      int l0 = labels[b * T_];
#pragma unroll
      for (int i = 0; i < R_; ++i)
#pragma unroll
        for (int j = 0; j < R_; ++j)
          res[i][j] = cb[i * (V_ * R_) + l0 * R_ + j];
    }
    for (int tt = 1; tt < T_; ++tt) {
      int l = labels[b * T_ + tt];
      float mt[R_][R_], nr[R_][R_];
#pragma unroll
      for (int i = 0; i < R_; ++i)
#pragma unroll
        for (int j = 0; j < R_; ++j)
          mt[i][j] = cb[i * (V_ * R_) + l * R_ + j];
#pragma unroll
      for (int i = 0; i < R_; ++i)
#pragma unroll
        for (int j = 0; j < R_; ++j) {
          float s = 0.f;
#pragma unroll
          for (int k = 0; k < R_; ++k) s += res[i][k] * mt[k][j];
          nr[i][j] = s;
        }
#pragma unroll
      for (int i = 0; i < R_; ++i)
#pragma unroll
        for (int j = 0; j < R_; ++j) res[i][j] = nr[i][j];
    }

    const float* al = alpha + b * R_;
    const float* be = beta + b * R_;
    float u = 0.f;
#pragma unroll
    for (int i = 0; i < R_; ++i)
#pragma unroll
      for (int j = 0; j < R_; ++j) u += al[i] * res[i][j] * be[j];

    // zmat = marg^(T+1)
    float zm[R_][R_];
#pragma unroll
    for (int i = 0; i < R_; ++i)
#pragma unroll
      for (int j = 0; j < R_; ++j) zm[i][j] = marg[i][j];
    for (int s = 0; s < T_; ++s) {
      float nr[R_][R_];
#pragma unroll
      for (int i = 0; i < R_; ++i)
#pragma unroll
        for (int j = 0; j < R_; ++j) {
          float acc2 = 0.f;
#pragma unroll
          for (int k = 0; k < R_; ++k) acc2 += zm[i][k] * marg[k][j];
          nr[i][j] = acc2;
        }
#pragma unroll
      for (int i = 0; i < R_; ++i)
#pragma unroll
        for (int j = 0; j < R_; ++j) zm[i][j] = nr[i][j];
    }
    float z = 0.f;
#pragma unroll
    for (int i = 0; i < R_; ++i)
#pragma unroll
      for (int j = 0; j < R_; ++j) z += al[i] * zm[i][j] * be[j];

    loss_b[b] = -logf(u / z);
  }
}

// ---------------- Kernel 5: mean over B into d_out ---------------------------
__global__ void final_kernel(const float* __restrict__ loss_b, float* __restrict__ out) {
  __shared__ float red[128];
  int t = threadIdx.x;
  red[t] = loss_b[t];
  __syncthreads();
  for (int s = 64; s > 0; s >>= 1) {
    if (t < s) red[t] += red[t + s];
    __syncthreads();
  }
  if (t == 0) out[0] = red[0] * (1.0f / B_);
}

extern "C" void kernel_launch(void* const* d_in, const int* in_sizes, int n_in,
                              void* d_out, int out_size, void* d_ws, size_t ws_size,
                              hipStream_t stream) {
  const float* input_embs = (const float*)d_in[0];   // [B,T,D] f32
  const int*   labels     = (const int*)d_in[1];     // [B,T] i32
  const float* wa         = (const float*)d_in[2];   // [D,R] f32
  const float* wb         = (const float*)d_in[3];   // [D,R] f32
  const float* wv         = (const float*)d_in[4];   // [D,V*R*R] f32
  float* out = (float*)d_out;

  char* ws = (char*)d_ws;
  float*    mean32 = (float*)(ws);                   // 1 MB
  _Float16* mean16 = (_Float16*)(ws + 1048576);      // 0.5 MB
  float*    core   = (float*)(ws + 1572864);         // 4 MB
  float*    alpha  = (float*)(ws + 5767168);         // 2 KB
  float*    beta   = (float*)(ws + 5769216);         // 2 KB
  float*    lossb  = (float*)(ws + 5771264);         // 512 B

  mean_embs_kernel<<<(B_ * D_ + 255) / 256, 256, 0, stream>>>(input_embs, mean32, mean16);
  alphabeta_kernel<<<B_, 256, 0, stream>>>(mean32, wa, wb, alpha, beta);
  core_gemm_kernel<<<N_ / BN_, 256, 0, stream>>>(mean16, wv, core);
  loss_kernel<<<B_, 256, 0, stream>>>(core, alpha, beta, labels, lossb);
  final_kernel<<<1, 128, 0, stream>>>(lossb, out);
}
